// qMoEModelBatched_88227218194501
// MI455X (gfx1250) — matrix operand order repaired and checked by reading, not compiled, not run on hardware
//
#include <hip/hip_runtime.h>
#include <hip/hip_bf16.h>

// ---------------------------------------------------------------------------
// Problem constants (from reference)
// ---------------------------------------------------------------------------
#define B_ROWS 16384
#define IN_DIM 768
#define RH     256
#define RH2    128
#define NEXP   8
#define NCLS   50
#define NCLS_P 64      // NCLS padded to block-tile N
#define RW3_P  64      // router W3 rows (8) padded to block-tile N
#define H1     512
#define H2     256
#define THR    0.05f

typedef _Float16 fh8   __attribute__((ext_vector_type(8)));
typedef _Float16 fh16  __attribute__((ext_vector_type(16)));
typedef float    f32x8 __attribute__((ext_vector_type(8)));

static __device__ __forceinline__ fh16 cat8(fh8 lo, fh8 hi) {
  return __builtin_shufflevector(lo, hi, 0,1,2,3,4,5,6,7,8,9,10,11,12,13,14,15);
}

// Unguarded fragment load: two b128 loads at q and q+gap (halves).
// A-fragment and B-fragment both gap=16 (K split 0..7 / 16..23 per lane half)
// per the CDNA5 WMMA VGPR layouts.
static __device__ __forceinline__ fh16 ldfrag(const _Float16* __restrict__ q, int gap) {
  fh8 lo = *(const fh8*)(q);
  fh8 hi = *(const fh8*)(q + gap);
  return cat8(lo, hi);
}

// ---------------------------------------------------------------------------
// f16 WMMA GEMM:  out = act( (A @ W^T) * scale + bias )
//   A: (M x K) f16 row-major, M multiple of 128.
//   W: (Npad x K) f16 row-major, Npad multiple of 64 (zero-padded rows).
//   Block = 256 threads = 8 waves; wave tile 32x32; block tile 128(M) x 64(N).
//   Software-pipelined K loop: next fragments load before current WMMAs.
//   gate != null: outF[m*N+n] += gate[m*gs+gc] * v  (per-expert combine)
//   else: write outH (f16) and/or outF (f32), ld = N (real).
// ---------------------------------------------------------------------------
__global__ __launch_bounds__(256)
void wmma_gemm_f16(const _Float16* __restrict__ A, const _Float16* __restrict__ W,
                   const float* __restrict__ bias, const float* __restrict__ scale,
                   const float* __restrict__ gate, int gs, int gc,
                   _Float16* __restrict__ outH, float* __restrict__ outF,
                   int M, int N, int K, int relu)
{
  const int lane = threadIdx.x & 31;
  const int wave = threadIdx.x >> 5;
  const int m0 = blockIdx.x * 128 + (wave & 3) * 32;
  const int n0 = blockIdx.y * 64  + (wave >> 2) * 32;

  // Per-lane base pointers (loop-invariant).
  const _Float16* pa0 = A + (size_t)(m0 + (lane & 15)) * K + ((lane >> 4) << 3);
  const _Float16* pa1 = pa0 + (size_t)16 * K;
  const _Float16* pb0 = W + (size_t)(n0 + (lane & 15)) * K + ((lane >> 4) << 3);
  const _Float16* pb1 = pb0 + (size_t)16 * K;

  f32x8 acc00 = {}, acc01 = {}, acc10 = {}, acc11 = {};

  fh16 a0 = ldfrag(pa0, 16);
  fh16 a1 = ldfrag(pa1, 16);
  fh16 b0 = ldfrag(pb0, 16);
  fh16 b1 = ldfrag(pb1, 16);

  for (int k = 32; k < K; k += 32) {
    fh16 na0 = ldfrag(pa0 + k, 16);
    fh16 na1 = ldfrag(pa1 + k, 16);
    fh16 nb0 = ldfrag(pb0 + k, 16);
    fh16 nb1 = ldfrag(pb1 + k, 16);
    acc00 = __builtin_amdgcn_wmma_f32_16x16x32_f16(false, a0, false, b0, (short)0, acc00, false, false);
    acc01 = __builtin_amdgcn_wmma_f32_16x16x32_f16(false, a0, false, b1, (short)0, acc01, false, false);
    acc10 = __builtin_amdgcn_wmma_f32_16x16x32_f16(false, a1, false, b0, (short)0, acc10, false, false);
    acc11 = __builtin_amdgcn_wmma_f32_16x16x32_f16(false, a1, false, b1, (short)0, acc11, false, false);
    a0 = na0; a1 = na1; b0 = nb0; b1 = nb1;
  }
  acc00 = __builtin_amdgcn_wmma_f32_16x16x32_f16(false, a0, false, b0, (short)0, acc00, false, false);
  acc01 = __builtin_amdgcn_wmma_f32_16x16x32_f16(false, a0, false, b1, (short)0, acc01, false, false);
  acc10 = __builtin_amdgcn_wmma_f32_16x16x32_f16(false, a1, false, b0, (short)0, acc10, false, false);
  acc11 = __builtin_amdgcn_wmma_f32_16x16x32_f16(false, a1, false, b1, (short)0, acc11, false, false);

  const float s = scale ? *scale : 1.0f;
  f32x8 accs[2][2] = { { acc00, acc01 }, { acc10, acc11 } };

  #pragma unroll
  for (int mi = 0; mi < 2; ++mi) {
    #pragma unroll
    for (int ni = 0; ni < 2; ++ni) {
      const int n = n0 + ni * 16 + (lane & 15);
      if (n < N) {
        const float bv = bias ? bias[n] : 0.0f;
        const int mb = m0 + mi * 16 + ((lane >> 4) << 3);
        #pragma unroll
        for (int r = 0; r < 8; ++r) {
          const int m = mb + r;
          float v = accs[mi][ni][r] * s + bv;
          if (relu) v = fmaxf(v, 0.0f);
          const size_t o = (size_t)m * N + n;
          if (gate) {
            outF[o] += gate[(size_t)m * gs + gc] * v;
          } else {
            if (outH) outH[o] = (_Float16)v;
            if (outF) outF[o] = v;
          }
        }
      }
    }
  }
}

// ---------------------------------------------------------------------------
// Pad-aware convert / ternarize:
//   out is (nexp x Npad x K) f16; in is (nexp x Nreal x K) f32.
//   Rows >= Nreal are written as zero.  tern!=0 applies BitNet ternarization
//   (sign(w) * (|w| > THR); per-matrix scale applied in the GEMM epilogue).
// ---------------------------------------------------------------------------
__global__ void cvt_pad_kernel(const float* __restrict__ in, _Float16* __restrict__ out,
                               int Nreal, int Npad, int K, int nexp, int tern, size_t total)
{
  size_t i = (size_t)blockIdx.x * blockDim.x + threadIdx.x;
  if (i >= total) return;
  float w = 0.0f;
  if (Nreal == Npad) {
    w = in[i];
  } else {
    const size_t perE = (size_t)Npad * K;
    const size_t e = i / perE;
    const size_t rem = i - e * perE;
    const int r = (int)(rem / K);
    const int c = (int)(rem - (size_t)r * K);
    if (r < Nreal) w = in[((size_t)e * Nreal + r) * K + c];
  }
  if (tern) w = (w > THR) ? 1.0f : ((w < -THR) ? -1.0f : 0.0f);
  out[i] = (_Float16)w;
}

// scale[e] = mean |W_e| ; one block per expert
__global__ __launch_bounds__(256)
void meanabs_kernel(const float* __restrict__ W, float* __restrict__ scale, int perExpert) {
  __shared__ float red[256];
  const float* p = W + (size_t)blockIdx.x * perExpert;
  float s = 0.0f;
  for (int i = threadIdx.x; i < perExpert; i += 256) s += fabsf(p[i]);
  red[threadIdx.x] = s;
  __syncthreads();
  for (int o = 128; o > 0; o >>= 1) {
    if ((int)threadIdx.x < o) red[threadIdx.x] += red[threadIdx.x + o];
    __syncthreads();
  }
  if (threadIdx.x == 0) scale[blockIdx.x] = red[0] / (float)perExpert;
}

// softmax over E=8 logits, write router_p, build dense gate = topk_val/TOPK
__global__ void router_softmax_topk(const float* __restrict__ logits,
                                    float* __restrict__ router_p,
                                    float* __restrict__ gate, int rows) {
  int b = blockIdx.x * blockDim.x + threadIdx.x;
  if (b >= rows) return;
  float l[NEXP];
  float mx = -3.4e38f;
  #pragma unroll
  for (int e = 0; e < NEXP; ++e) { l[e] = logits[(size_t)b * NEXP + e]; mx = fmaxf(mx, l[e]); }
  float sum = 0.0f;
  #pragma unroll
  for (int e = 0; e < NEXP; ++e) { l[e] = __expf(l[e] - mx); sum += l[e]; }
  const float inv = 1.0f / sum;
  #pragma unroll
  for (int e = 0; e < NEXP; ++e) { l[e] *= inv; router_p[(size_t)b * NEXP + e] = l[e]; }
  int i1 = 0; float v1 = l[0];
  #pragma unroll
  for (int e = 1; e < NEXP; ++e) if (l[e] > v1) { v1 = l[e]; i1 = e; }
  int i2 = -1; float v2 = -3.4e38f;
  #pragma unroll
  for (int e = 0; e < NEXP; ++e) if (e != i1 && l[e] > v2) { v2 = l[e]; i2 = e; }
  #pragma unroll
  for (int e = 0; e < NEXP; ++e) {
    float g = (e == i1) ? v1 * 0.5f : ((e == i2) ? v2 * 0.5f : 0.0f);
    gate[(size_t)b * NEXP + e] = g;
  }
}

__global__ void zero_out_kernel(float* __restrict__ out, int n, int lbIdx) {
  int i = blockIdx.x * blockDim.x + threadIdx.x;
  if (i < n) out[i] = 0.0f;
  if (i == 0) out[lbIdx] = 0.0f;   // lb_loss = 0 (eval mode)
}

// ---------------------------------------------------------------------------
// Launch
// ---------------------------------------------------------------------------
extern "C" void kernel_launch(void* const* d_in, const int* in_sizes, int n_in,
                              void* d_out, int out_size, void* d_ws, size_t ws_size,
                              hipStream_t stream) {
  (void)in_sizes; (void)n_in; (void)out_size; (void)ws_size;

  const float* x   = (const float*)d_in[0];
  const float* rW1 = (const float*)d_in[1];  const float* rb1 = (const float*)d_in[2];
  const float* rW2 = (const float*)d_in[3];  const float* rb2 = (const float*)d_in[4];
  const float* rW3 = (const float*)d_in[5];  const float* rb3 = (const float*)d_in[6];
  const float* eW1 = (const float*)d_in[7];  const float* eb1 = (const float*)d_in[8];
  const float* eW2 = (const float*)d_in[9];  const float* eb2 = (const float*)d_in[10];
  const float* eW3 = (const float*)d_in[11]; const float* eb3 = (const float*)d_in[12];

  float* outF  = (float*)d_out;                 // [B*C] combined logits
  float* routP = outF + (size_t)B_ROWS * NCLS;  // [B*E] router probabilities
  const int lbIdx = B_ROWS * NCLS + B_ROWS * NEXP;

  // ---- workspace carve-out (256B aligned) ----
  uint8_t* ws = (uint8_t*)d_ws;
  size_t off = 0;
  auto carve = [&](size_t bytes) -> void* {
    void* p = ws + off;
    off = (off + bytes + 255) & ~(size_t)255;
    return p;
  };
  _Float16* xh    = (_Float16*)carve((size_t)B_ROWS * IN_DIM * 2);
  _Float16* rW1h  = (_Float16*)carve((size_t)RH * IN_DIM * 2);
  _Float16* rW2h  = (_Float16*)carve((size_t)RH2 * RH * 2);
  _Float16* rW3h  = (_Float16*)carve((size_t)RW3_P * RH2 * 2);          // padded 64 x 128
  _Float16* eW1q  = (_Float16*)carve((size_t)NEXP * H1 * IN_DIM * 2);
  _Float16* eW2q  = (_Float16*)carve((size_t)NEXP * H2 * H1 * 2);
  _Float16* eW3q  = (_Float16*)carve((size_t)NEXP * NCLS_P * H2 * 2);   // padded E x 64 x 256
  float*    s1    = (float*)carve(NEXP * 4);
  float*    s2    = (float*)carve(NEXP * 4);
  float*    s3    = (float*)carve(NEXP * 4);
  _Float16* h1h   = (_Float16*)carve((size_t)B_ROWS * RH * 2);
  _Float16* h2h   = (_Float16*)carve((size_t)B_ROWS * RH2 * 2);
  float*    logit = (float*)carve((size_t)B_ROWS * NEXP * 4);
  float*    gate  = (float*)carve((size_t)B_ROWS * NEXP * 4);
  _Float16* a1h   = (_Float16*)carve((size_t)B_ROWS * H1 * 2);
  _Float16* a2h   = (_Float16*)carve((size_t)B_ROWS * H2 * 2);

  const dim3 blk(256);
  auto nb = [](size_t n) { return (unsigned)((n + 255) / 256); };

  auto cvt = [&](const float* in, _Float16* out, int Nreal, int Npad, int K, int nexp, int tern) {
    size_t total = (size_t)nexp * Npad * K;
    cvt_pad_kernel<<<nb(total), blk, 0, stream>>>(in, out, Nreal, Npad, K, nexp, tern, total);
  };

  // ---- convert fp32 -> f16 (pad odd-shaped weights to tile-aligned rows) ----
  cvt(x,   xh,   B_ROWS, B_ROWS, IN_DIM, 1, 0);
  cvt(rW1, rW1h, RH,     RH,     IN_DIM, 1, 0);
  cvt(rW2, rW2h, RH2,    RH2,    RH,     1, 0);
  cvt(rW3, rW3h, NEXP,   RW3_P,  RH2,    1, 0);

  // ---- per-expert mean|W| scales + ternarize (ternary exact in f16) ----
  meanabs_kernel<<<NEXP, blk, 0, stream>>>(eW1, s1, H1 * IN_DIM);
  meanabs_kernel<<<NEXP, blk, 0, stream>>>(eW2, s2, H2 * H1);
  meanabs_kernel<<<NEXP, blk, 0, stream>>>(eW3, s3, NCLS * H2);
  cvt(eW1, eW1q, H1,   H1,     IN_DIM, NEXP, 1);
  cvt(eW2, eW2q, H2,   H2,     H1,     NEXP, 1);
  cvt(eW3, eW3q, NCLS, NCLS_P, H2,     NEXP, 1);

  auto gemm = [&](const _Float16* A, const _Float16* W, const float* bias, const float* scale,
                  const float* g, int gc, _Float16* oh, float* of,
                  int M, int Nreal, int Npad, int K, int relu) {
    dim3 grid((unsigned)(M / 128), (unsigned)(Npad / 64));
    wmma_gemm_f16<<<grid, blk, 0, stream>>>(A, W, bias, scale, g, NEXP, gc, oh, of, M, Nreal, K, relu);
  };

  // ---- router MLP: 768 -> 256 -> 128 -> 8 ----
  gemm(xh,  rW1h, rb1, nullptr, nullptr, 0, h1h, nullptr, B_ROWS, RH,   RH,    IN_DIM, 1);
  gemm(h1h, rW2h, rb2, nullptr, nullptr, 0, h2h, nullptr, B_ROWS, RH2,  RH2,   RH,     1);
  gemm(h2h, rW3h, rb3, nullptr, nullptr, 0, nullptr, logit, B_ROWS, NEXP, RW3_P, RH2,  0);

  // ---- softmax + top-2 gating; zero the combined output ----
  router_softmax_topk<<<nb(B_ROWS), blk, 0, stream>>>(logit, routP, gate, B_ROWS);
  zero_out_kernel<<<nb((size_t)B_ROWS * NCLS), blk, 0, stream>>>(outF, B_ROWS * NCLS, lbIdx);

  // ---- experts: 768 -> 512 -> 256 -> 50, gated accumulate into d_out ----
  for (int e = 0; e < NEXP; ++e) {
    gemm(xh,  eW1q + (size_t)e * H1 * IN_DIM, eb1 + (size_t)e * H1, s1 + e,
         nullptr, 0, a1h, nullptr, B_ROWS, H1, H1, IN_DIM, 1);
    gemm(a1h, eW2q + (size_t)e * H2 * H1,     eb2 + (size_t)e * H2, s2 + e,
         nullptr, 0, a2h, nullptr, B_ROWS, H2, H2, H1, 1);
    gemm(a2h, eW3q + (size_t)e * NCLS_P * H2, eb3 + (size_t)e * NCLS, s3 + e,
         gate, e, nullptr, outF, B_ROWS, NCLS, NCLS_P, H2, 0);
  }
}
